// EquivariantDiffusionModel_90692529422958
// MI455X (gfx1250) — matrix operand order repaired and checked by reading, not compiled, not run on hardware
//
#include <hip/hip_runtime.h>
#include <math.h>
#include <stdint.h>

typedef __attribute__((ext_vector_type(16))) __bf16 v16bf;
typedef __attribute__((ext_vector_type(8)))  float  v8f;

#if defined(__has_builtin)
# if __has_builtin(__builtin_amdgcn_tensor_load_to_lds) && \
     __has_builtin(__builtin_amdgcn_s_wait_tensorcnt)
#  define USE_TDM 1
# endif
#endif
#ifndef USE_TDM
# define USE_TDM 0
#endif

static __device__ __forceinline__ float silu_f(float x) {
    return x / (1.0f + __expf(-x));
}

#if USE_TDM
typedef unsigned int u32x4 __attribute__((ext_vector_type(4)));
typedef int          i32x4 __attribute__((ext_vector_type(4)));
typedef int          i32x8 __attribute__((ext_vector_type(8)));

// LDS byte address: flat-address low 32 bits are the LDS byte address
// (ISA 10.2 aperture mapping).
static __device__ __forceinline__ unsigned lds_byte(const void* p) {
    return (unsigned)(uintptr_t)p;
}

// Issue one TDM 2-D tile load: rows x 32 bf16 tile from a [*, stride] tensor.
// rows_rem bounds dim1 so OOB rows are hardware zero-filled (ISA 8.2).
static __device__ __forceinline__ void tdm_load_tile(
    const __bf16* gsrc, unsigned lds_addr,
    unsigned rows_rem, unsigned tile_rows, unsigned k_rem, unsigned stride)
{
    const unsigned long long ga = (unsigned long long)(uintptr_t)gsrc;
    u32x4 g0;
    g0[0] = 1u;                                   // count=1, user desc, no gather
    g0[1] = lds_addr;                             // lds_addr (bytes)
    g0[2] = (unsigned)ga;                         // global_addr[31:0]
    g0[3] = (unsigned)((ga >> 32) & 0x01FFFFFFu)  // global_addr[56:32]
          | (2u << 30);                           // type = 2 ("image")
    i32x8 g1;
    g1[0] = (int)(1u << 16);                      // data_size = 1 -> 2 bytes
    g1[1] = (int)((k_rem & 0xFFFFu) << 16);       // tensor_dim0[15:0]
    g1[2] = (int)((k_rem >> 16) | ((rows_rem & 0xFFFFu) << 16)); // dim0 hi | dim1 lo
    g1[3] = (int)((rows_rem >> 16) | (32u << 16));               // dim1 hi | tile_dim0=32
    g1[4] = (int)(tile_rows & 0xFFFFu);           // tile_dim1 (tile_dim2 = 0)
    g1[5] = (int)stride;                          // tensor_dim0_stride[31:0]
    g1[6] = 0;                                    // stride hi | dim1_stride lo
    g1[7] = 0;
    i32x4 z4 = {0, 0, 0, 0};
#if defined(__clang_major__) && __clang_major__ >= 23
    i32x8 z8 = {0, 0, 0, 0, 0, 0, 0, 0};
    __builtin_amdgcn_tensor_load_to_lds(g0, g1, z4, z4, z8, 0);
#else
    __builtin_amdgcn_tensor_load_to_lds(g0, g1, z4, z4, 0);
#endif
}
#endif // USE_TDM

// ---------------------------------------------------------------------------
// bf16 WMMA GEMM:  out = act(A[M,K] @ B[K,N] + bias[N])
// A row-major (lda). B supplied TRANSPOSED: Bt[N][K] (ldbt == K) so both LDS
// tiles are read as contiguous 16B fragment loads. Macro-tile 128x64, 8 waves
// (4x2), each wave a 32x32 quad of 16x16 WMMA tiles. Double-buffered LDS
// k-pipeline; tiles staged by the Tensor Data Mover when available.
// ---------------------------------------------------------------------------
__launch_bounds__(256)
__global__ void gemm_bf16_kernel(const __bf16* __restrict__ A, int lda,
                                 const __bf16* __restrict__ Bt,
                                 const float* __restrict__ bias,
                                 int M, int K, int act,
                                 __bf16* __restrict__ out1, int ld1, int off1,
                                 __bf16* __restrict__ out2, int ld2, int off2)
{
    __shared__ __bf16 As[2][128 * 32];   // [m][k] tile
    __shared__ __bf16 Bs[2][64 * 32];    // [n][k] tile (B pre-transposed)

    const int tid  = threadIdx.x;
    const int lane = tid & 31;
    const int wave = tid >> 5;
    const int mt   = wave >> 1;          // 0..3
    const int nt   = wave & 1;           // 0..1
    const int m0   = blockIdx.x * 128;
    const int n0   = blockIdx.y * 64;

#if USE_TDM
    // ---- prologue: DMA k=0 tiles via Tensor Data Mover (wave 0 only) ----
    if (wave == 0) {
        tdm_load_tile(A  + (size_t)m0 * lda, lds_byte(&As[0][0]),
                      (unsigned)(M - m0), 128u, (unsigned)K, (unsigned)lda);
        tdm_load_tile(Bt + (size_t)n0 * K,   lds_byte(&Bs[0][0]),
                      64u, 64u, (unsigned)K, (unsigned)K);
        __builtin_amdgcn_s_wait_tensorcnt(0);
    }
    __syncthreads();
#else
    const int srow = tid >> 2;           // 0..63
    const int scol = (tid & 3) * 8;      // 0,8,16,24
    const float4 zero4 = make_float4(0.f, 0.f, 0.f, 0.f);
    float4 av0 = zero4, av1 = zero4, bv;
    {
        const int r0 = m0 + srow, r1 = m0 + 64 + srow;
        if (r0 < M) av0 = *(const float4*)(A + (size_t)r0 * lda + scol);
        if (r1 < M) av1 = *(const float4*)(A + (size_t)r1 * lda + scol);
        bv = *(const float4*)(Bt + (size_t)(n0 + srow) * K + scol);
    }
    *(float4*)(&As[0][srow * 32 + scol])        = av0;
    *(float4*)(&As[0][(64 + srow) * 32 + scol]) = av1;
    *(float4*)(&Bs[0][srow * 32 + scol])        = bv;
    __syncthreads();
#endif

    v8f acc[2][2] = {};
    int buf = 0;

    for (int k0 = 0; k0 < K; k0 += 32) {
        const bool more = (k0 + 32) < K;
        const int kn = k0 + 32;
#if USE_TDM
        if (more && wave == 0) {   // DMA next tiles while we compute this one
            tdm_load_tile(A  + (size_t)m0 * lda + kn, lds_byte(&As[buf ^ 1][0]),
                          (unsigned)(M - m0), 128u, (unsigned)(K - kn), (unsigned)lda);
            tdm_load_tile(Bt + (size_t)n0 * K + kn,   lds_byte(&Bs[buf ^ 1][0]),
                          64u, 64u, (unsigned)(K - kn), (unsigned)K);
        }
#else
        if (more) {
            const int r0 = m0 + srow, r1 = m0 + 64 + srow;
            av0 = zero4; av1 = zero4;
            if (r0 < M) av0 = *(const float4*)(A + (size_t)r0 * lda + kn + scol);
            if (r1 < M) av1 = *(const float4*)(A + (size_t)r1 * lda + kn + scol);
            bv = *(const float4*)(Bt + (size_t)(n0 + srow) * K + kn + scol);
            if (k0 + 64 < K)
                __builtin_prefetch(A + (size_t)(m0 + srow) * lda + k0 + 64 + scol, 0, 0);
        }
#endif

        // ---- fragments (ISA 7.12.2 wave32 layouts; all 16B LDS loads) ----
        const int r  = lane & 15;
        const int hi = lane >> 4;
        union { v16bf v; float4 f[2]; } a[2], b[2];
        #pragma unroll
        for (int im = 0; im < 2; ++im) {
            const __bf16* p = &As[buf][(mt * 32 + im * 16 + r) * 32 + hi * 8];
            a[im].f[0] = *(const float4*)p;        // K = hi*8 .. +7
            a[im].f[1] = *(const float4*)(p + 16); // K = hi*8+16 .. +23
        }
        #pragma unroll
        for (int in = 0; in < 2; ++in) {
            const __bf16* p = &Bs[buf][(nt * 32 + in * 16 + r) * 32 + hi * 8];
            b[in].f[0] = *(const float4*)p;        // K = hi*8 .. +7
            b[in].f[1] = *(const float4*)(p + 16); // K = hi*8+16 .. +23
        }

        #pragma unroll
        for (int im = 0; im < 2; ++im)
            #pragma unroll
            for (int in = 0; in < 2; ++in)
                acc[im][in] = __builtin_amdgcn_wmma_f32_16x16x32_bf16(
                    false, a[im].v, false, b[in].v, (short)0, acc[im][in],
                    false, false);

        if (more) {
#if USE_TDM
            if (wave == 0) __builtin_amdgcn_s_wait_tensorcnt(0);
            __syncthreads();
#else
            const int nb = buf ^ 1;
            *(float4*)(&As[nb][srow * 32 + scol])        = av0;
            *(float4*)(&As[nb][(64 + srow) * 32 + scol]) = av1;
            *(float4*)(&Bs[nb][srow * 32 + scol])        = bv;
            __syncthreads();
#endif
            buf ^= 1;
        }
    }

    // ---- epilogue: bias + (SiLU) + bf16 store(s) ----
    const int cc = lane & 15;
    #pragma unroll
    for (int im = 0; im < 2; ++im) {
        const int rbase = m0 + mt * 32 + im * 16 + (lane >> 4) * 8;
        #pragma unroll
        for (int in = 0; in < 2; ++in) {
            const int col = n0 + nt * 32 + in * 16 + cc;
            const float bc = bias ? bias[col] : 0.0f;
            #pragma unroll
            for (int v = 0; v < 8; ++v) {
                const int row = rbase + v;
                if (row < M) {
                    float x = acc[im][in][v] + bc;
                    if (act) x = silu_f(x);
                    const __bf16 xb = (__bf16)x;
                    out1[(size_t)row * ld1 + off1 + col] = xb;
                    if (out2) out2[(size_t)row * ld2 + off2 + col] = xb;
                }
            }
        }
    }
}

// ---------------------------------------------------------------------------
// weight convert+transpose: dst[l][n][k] = (bf16) src[l][k][n]
__global__ void wtrans_kernel(const float* __restrict__ src,
                              __bf16* __restrict__ dst,
                              int K, int N, long long total)
{
    long long tid = (long long)blockIdx.x * blockDim.x + threadIdx.x;
    if (tid >= total) return;
    const long long per = (long long)K * N;
    const int l = (int)(tid / per);
    const int rem = (int)(tid % per);
    const int n = rem / K, k = rem % K;
    dst[tid] = (__bf16)src[(long long)l * per + (long long)k * N + n];
}

__global__ void f32_to_bf16_kernel(const float* __restrict__ src,
                                   __bf16* __restrict__ dst, int n) {
    int i = blockIdx.x * blockDim.x + threadIdx.x;
    if (i < n) dst[i] = (__bf16)src[i];
}

// timestep embedding + 2-layer MLP, B<=16 rows — single block, 256 threads
__launch_bounds__(256)
__global__ void temb_kernel(const int* __restrict__ t,
                            const float* __restrict__ w1, const float* __restrict__ b1,
                            const float* __restrict__ w2, const float* __restrict__ b2,
                            float* __restrict__ temb, int Bn)
{
    __shared__ float te[16 * 256];
    __shared__ float hid[16 * 256];
    const int c = threadIdx.x;
    const int d = (c < 128) ? c : (c - 128);
    const float f = __expf(-9.210340371976184f * (float)d / 127.0f);
    for (int b = 0; b < Bn; ++b) {
        float ang = (float)t[b] * f;
        te[b * 256 + c] = (c < 128) ? sinf(ang) : cosf(ang);
    }
    __syncthreads();
    for (int b = 0; b < Bn; ++b) {
        float s = b1[c];
        for (int k = 0; k < 256; ++k) s += te[b * 256 + k] * w1[k * 256 + c];
        hid[b * 256 + c] = silu_f(s);
    }
    __syncthreads();
    for (int b = 0; b < Bn; ++b) {
        float s = b2[c];
        for (int k = 0; k < 256; ++k) s += hid[b * 256 + k] * w2[k * 256 + c];
        temb[b * 256 + c] = s;
    }
}

__global__ void h_init_kernel(const int* __restrict__ atom_type,
                              const int* __restrict__ batch,
                              const float* __restrict__ emb_table,
                              const float* __restrict__ temb,
                              float* __restrict__ h, int N) {
    int idx = blockIdx.x * blockDim.x + threadIdx.x;
    if (idx >= N * 256) return;
    int i = idx >> 8, c = idx & 255;
    h[idx] = emb_table[atom_type[i] * 256 + c] + temb[batch[i] * 256 + c];
}

// edge encoder: d0 -> [1,64] -> silu -> [64,64]; one 64-thread block per edge
__launch_bounds__(64)
__global__ void edge_enc_kernel(const float* __restrict__ pos,
                                const int* __restrict__ ei, int E,
                                const float* __restrict__ w1, const float* __restrict__ b1,
                                const float* __restrict__ w2, const float* __restrict__ b2,
                                __bf16* __restrict__ edge_attr)
{
    __shared__ float hid[64];
    const int e = blockIdx.x;
    const int c = threadIdx.x;
    const int s = ei[e], d = ei[E + e];
    const float dx = pos[s * 3 + 0] - pos[d * 3 + 0];
    const float dy = pos[s * 3 + 1] - pos[d * 3 + 1];
    const float dz = pos[s * 3 + 2] - pos[d * 3 + 2];
    const float d0 = sqrtf(dx * dx + dy * dy + dz * dz);
    hid[c] = silu_f(d0 * w1[c] + b1[c]);
    __syncthreads();
    float sum = b2[c];
    for (int k = 0; k < 64; ++k) sum += hid[k] * w2[k * 64 + c];
    edge_attr[(size_t)e * 64 + c] = (__bf16)sum;
}

// build xcat[e] = [ h[dst] | h[src] | edge_attr[e] ] as bf16, 4 cols/thread
__global__ void gather_kernel(const float* __restrict__ h,
                              const __bf16* __restrict__ edge_attr,
                              const int* __restrict__ ei, int E,
                              __bf16* __restrict__ xcat)
{
    long long tid = (long long)blockIdx.x * blockDim.x + threadIdx.x;
    if (tid >= (long long)E * 144) return;
    int e = (int)(tid / 144);
    int col = (int)(tid % 144) * 4;
    __bf16* o = xcat + (size_t)e * 576 + col;
    if (col < 256) {
        const float* hp = h + (size_t)ei[E + e] * 256 + col;      // x_i = h[dst]
        for (int i = 0; i < 4; ++i) o[i] = (__bf16)hp[i];
    } else if (col < 512) {
        const float* hp = h + (size_t)ei[e] * 256 + (col - 256);  // x_j = h[src]
        for (int i = 0; i < 4; ++i) o[i] = (__bf16)hp[i];
    } else {
        const __bf16* ap = edge_attr + (size_t)e * 64 + (col - 512);
        for (int i = 0; i < 4; ++i) o[i] = ap[i];
    }
}

// cw[e] = dot(cw_hidden[e,0:256], w2[0:256]) — wave per edge, shuffle reduce
__launch_bounds__(256)
__global__ void cw_dot_kernel(const __bf16* __restrict__ hid,
                              const float* __restrict__ w2,
                              float* __restrict__ cw, int E)
{
    int gw   = (int)((blockIdx.x * blockDim.x + threadIdx.x) >> 5);
    int lane = threadIdx.x & 31;
    if (gw >= E) return;
    float s = 0.f;
    #pragma unroll
    for (int j = 0; j < 8; ++j) {
        int c = lane + j * 32;
        s += (float)hid[(size_t)gw * 256 + c] * w2[c];
    }
    for (int off = 16; off > 0; off >>= 1) s += __shfl_down(s, off, 32);
    if (lane == 0) cw[gw] = s;
}

// scatter: h[dst] += nm ; posbuf[dst] += cw * rel/(|rel|+1e-8). Wave per edge.
__launch_bounds__(256)
__global__ void scatter_kernel(const __bf16* __restrict__ nm,   // cwcat, ld=320
                               const float* __restrict__ cw,
                               const float* __restrict__ posold,
                               const int* __restrict__ ei, int E,
                               float* __restrict__ h,
                               float* __restrict__ posbuf)
{
    int gw   = (int)((blockIdx.x * blockDim.x + threadIdx.x) >> 5);
    int lane = threadIdx.x & 31;
    if (gw >= E) return;
    const int src = ei[gw], dst = ei[E + gw];
    #pragma unroll
    for (int j = 0; j < 8; ++j) {
        int c = lane + j * 32;
        float v = (float)nm[(size_t)gw * 320 + c];
        __hip_atomic_fetch_add(&h[(size_t)dst * 256 + c], v,
                               __ATOMIC_RELAXED, __HIP_MEMORY_SCOPE_AGENT);
    }
    if (lane < 3) {
        float rx = posold[src * 3 + 0] - posold[dst * 3 + 0];
        float ry = posold[src * 3 + 1] - posold[dst * 3 + 1];
        float rz = posold[src * 3 + 2] - posold[dst * 3 + 2];
        float dist = sqrtf(rx * rx + ry * ry + rz * rz) + 1e-8f;
        float r = (lane == 0) ? rx : ((lane == 1) ? ry : rz);
        __hip_atomic_fetch_add(&posbuf[dst * 3 + lane], cw[gw] * r / dist,
                               __ATOMIC_RELAXED, __HIP_MEMORY_SCOPE_AGENT);
    }
}

// narrow output head: out[i,c] = b2[c] + hid[i,:]·w2[:,c], OC in {3,11}
__global__ void head2_kernel(const __bf16* __restrict__ hid,
                             const float* __restrict__ w2,
                             const float* __restrict__ b2,
                             float* __restrict__ out, int N, int OC)
{
    int idx = blockIdx.x * blockDim.x + threadIdx.x;
    if (idx >= N * OC) return;
    int i = idx / OC, c = idx % OC;
    float s = b2[c];
    for (int k = 0; k < 256; ++k)
        s += (float)hid[(size_t)i * 256 + k] * w2[k * OC + c];
    out[idx] = s;
}

// ---------------------------------------------------------------------------
extern "C" void kernel_launch(void* const* d_in, const int* in_sizes, int n_in,
                              void* d_out, int out_size, void* d_ws, size_t ws_size,
                              hipStream_t stream)
{
    const int*   atom_type = (const int*)  d_in[0];
    const float* pos_in    = (const float*)d_in[1];
    const int*   t_in      = (const int*)  d_in[2];
    const int*   batch     = (const int*)  d_in[3];
    const int*   ei        = (const int*)  d_in[4];
    const float* emb_table = (const float*)d_in[5];
    const float* time_w1   = (const float*)d_in[6];
    const float* time_b1   = (const float*)d_in[7];
    const float* time_w2   = (const float*)d_in[8];
    const float* time_b2   = (const float*)d_in[9];
    const float* eenc_w1   = (const float*)d_in[10];
    const float* eenc_b1   = (const float*)d_in[11];
    const float* eenc_w2   = (const float*)d_in[12];
    const float* eenc_b2   = (const float*)d_in[13];
    const float* node_w1   = (const float*)d_in[14];
    const float* node_b1   = (const float*)d_in[15];
    const float* node_w2   = (const float*)d_in[16];
    const float* node_b2   = (const float*)d_in[17];
    const float* edge_w1   = (const float*)d_in[18];
    const float* edge_b1   = (const float*)d_in[19];
    const float* edge_w2   = (const float*)d_in[20];
    const float* edge_b2   = (const float*)d_in[21];
    const float* coord_w1  = (const float*)d_in[22];
    const float* coord_b1  = (const float*)d_in[23];
    const float* coord_w2  = (const float*)d_in[24];
    const float* posout_w1 = (const float*)d_in[25];
    const float* posout_b1 = (const float*)d_in[26];
    const float* posout_w2 = (const float*)d_in[27];
    const float* posout_b2 = (const float*)d_in[28];
    const float* atomout_w1= (const float*)d_in[29];
    const float* atomout_b1= (const float*)d_in[30];
    const float* atomout_w2= (const float*)d_in[31];
    const float* atomout_b2= (const float*)d_in[32];

    const int N  = in_sizes[0];
    const int E  = in_sizes[4] / 2;
    const int Bn = in_sizes[2];
    const int L  = 9;

    // ---- workspace bump allocator ----
    char* wp = (char*)d_ws;
    auto alloc = [&](size_t bytes) {
        char* p = wp;
        wp += (bytes + 255) & ~(size_t)255;
        return p;
    };
    // transposed bf16 weights: Bt[l][n][k]
    __bf16* node_w1_t  = (__bf16*)alloc((size_t)L * 576 * 512 * 2);
    __bf16* node_w2_t  = (__bf16*)alloc((size_t)L * 512 * 256 * 2);
    __bf16* edge_w1_t  = (__bf16*)alloc((size_t)L * 576 * 64 * 2);
    __bf16* edge_w2_t  = (__bf16*)alloc((size_t)L * 64 * 64 * 2);
    __bf16* coord_w1_t = (__bf16*)alloc((size_t)L * 320 * 256 * 2);
    __bf16* pow1_t     = (__bf16*)alloc((size_t)256 * 256 * 2);
    __bf16* aow1_t     = (__bf16*)alloc((size_t)256 * 256 * 2);
    __bf16* xcat       = (__bf16*)alloc((size_t)E * 576 * 2);
    __bf16* buf1       = (__bf16*)alloc((size_t)E * 512 * 2);   // ef_hid / nm_hid / cw_hid
    __bf16* cwcat      = (__bf16*)alloc((size_t)E * 320 * 2);   // [nm | ef]
    __bf16* edge_attr  = (__bf16*)alloc((size_t)E * 64 * 2);
    float*  cwv        = (float*) alloc((size_t)E * 4);
    float*  h          = (float*) alloc((size_t)N * 256 * 4);
    float*  posbuf     = (float*) alloc((size_t)N * 3 * 4);
    float*  posold     = (float*) alloc((size_t)N * 3 * 4);
    float*  temb       = (float*) alloc((size_t)16 * 256 * 4);
    __bf16* hbf        = (__bf16*)alloc((size_t)N * 256 * 2);
    __bf16* headhid    = (__bf16*)alloc((size_t)N * 256 * 2);

    auto wtrans = [&](const float* s, __bf16* d, int K, int Nn, int Ln) {
        long long total = (long long)Ln * K * Nn;
        wtrans_kernel<<<(int)((total + 255) / 256), 256, 0, stream>>>(s, d, K, Nn, total);
    };
    wtrans(node_w1,   node_w1_t,  576, 512, L);
    wtrans(node_w2,   node_w2_t,  512, 256, L);
    wtrans(edge_w1,   edge_w1_t,  576, 64,  L);
    wtrans(edge_w2,   edge_w2_t,  64,  64,  L);
    wtrans(coord_w1,  coord_w1_t, 320, 256, L);
    wtrans(posout_w1, pow1_t,     256, 256, 1);
    wtrans(atomout_w1, aow1_t,    256, 256, 1);

    auto gemm = [&](const __bf16* A, int lda, const __bf16* Bt, int Nn,
                    const float* bias, int M, int K, int act,
                    __bf16* o1, int l1, int of1,
                    __bf16* o2, int l2, int of2) {
        dim3 g((M + 127) / 128, Nn / 64);
        gemm_bf16_kernel<<<g, 256, 0, stream>>>(A, lda, Bt, bias, M, K, act,
                                                o1, l1, of1, o2, l2, of2);
    };

    // ---- initial node features & edge attributes ----
    temb_kernel<<<1, 256, 0, stream>>>(t_in, time_w1, time_b1, time_w2, time_b2, temb, Bn);
    h_init_kernel<<<(N * 256 + 255) / 256, 256, 0, stream>>>(atom_type, batch, emb_table, temb, h, N);
    hipMemcpyAsync(posbuf, pos_in, (size_t)N * 3 * 4, hipMemcpyDeviceToDevice, stream);
    edge_enc_kernel<<<E, 64, 0, stream>>>(pos_in, ei, E, eenc_w1, eenc_b1, eenc_w2, eenc_b2, edge_attr);

    const int gwaves = (E * 32 + 255) / 256;
    for (int l = 0; l < L; ++l) {
        const __bf16* ew1 = edge_w1_t  + (size_t)l * 576 * 64;
        const __bf16* ew2 = edge_w2_t  + (size_t)l * 64 * 64;
        const __bf16* nw1 = node_w1_t  + (size_t)l * 576 * 512;
        const __bf16* nw2 = node_w2_t  + (size_t)l * 512 * 256;
        const __bf16* cw1 = coord_w1_t + (size_t)l * 320 * 256;
        const float*  eb1 = edge_b1 + l * 64;
        const float*  eb2 = edge_b2 + l * 64;
        const float*  nb1 = node_b1 + l * 512;
        const float*  nb2 = node_b2 + l * 256;
        const float*  cb1 = coord_b1 + l * 256;
        const float*  cw2 = coord_w2 + l * 256;

        // xcat = [h[dst] | h[src] | edge_attr]
        long long gt = (long long)E * 144;
        gather_kernel<<<(int)((gt + 255) / 256), 256, 0, stream>>>(h, edge_attr, ei, E, xcat);

        // ef = mlp2(xcat) ; write into xcat[:,512:576] and cwcat[:,256:320]
        gemm(xcat, 576, ew1, 64, eb1, E, 576, 1, buf1, 64, 0, nullptr, 0, 0);
        gemm(buf1, 64, ew2, 64, eb2, E, 64, 0, xcat, 576, 512, cwcat, 320, 256);

        // nm = mlp2([x_i|x_j|ef]) ; write into cwcat[:,0:256]
        gemm(xcat, 576, nw1, 512, nb1, E, 576, 1, buf1, 512, 0, nullptr, 0, 0);
        gemm(buf1, 512, nw2, 256, nb2, E, 512, 0, cwcat, 320, 0, nullptr, 0, 0);

        // cw = silu([nm|ef] @ cw1 + cb1) @ cw2
        gemm(cwcat, 320, cw1, 256, cb1, E, 320, 1, buf1, 256, 0, nullptr, 0, 0);
        cw_dot_kernel<<<gwaves, 256, 0, stream>>>(buf1, cw2, cwv, E);

        // h += segsum(nm); pos += segsum(cw * rel/(|rel|+eps))
        hipMemcpyAsync(posold, posbuf, (size_t)N * 3 * 4, hipMemcpyDeviceToDevice, stream);
        scatter_kernel<<<gwaves, 256, 0, stream>>>(cwcat, cwv, posold, ei, E, h, posbuf);
    }

    // ---- output heads ----
    f32_to_bf16_kernel<<<(N * 256 + 255) / 256, 256, 0, stream>>>(h, hbf, N * 256);
    gemm(hbf, 256, pow1_t, 256, posout_b1, N, 256, 1, headhid, 256, 0, nullptr, 0, 0);
    head2_kernel<<<(N * 3 + 255) / 256, 256, 0, stream>>>(headhid, posout_w2, posout_b2,
                                                          (float*)d_out, N, 3);
    gemm(hbf, 256, aow1_t, 256, atomout_b1, N, 256, 1, headhid, 256, 0, nullptr, 0, 0);
    head2_kernel<<<(N * 11 + 255) / 256, 256, 0, stream>>>(headhid, atomout_w2, atomout_b2,
                                                           (float*)d_out + (size_t)N * 3, N, 11);

    (void)n_in; (void)out_size; (void)ws_size;
}
